// AsymFormer_51642686767352
// MI455X (gfx1250) — matrix operand order repaired and checked by reading, not compiled, not run on hardware
//
#include <hip/hip_runtime.h>

// ---------------------------------------------------------------------------
// AsymFormer forward for MI455X (gfx1250, wave32, WMMA f32_16x16x32_f16,
// TDM tensor_load_to_lds for the dominant f16 activation GEMMs)
// ---------------------------------------------------------------------------

typedef _Float16 half_t;
typedef __attribute__((ext_vector_type(16))) _Float16 v16h;
typedef __attribute__((ext_vector_type(8)))  float    v8f;
typedef unsigned int v4u __attribute__((ext_vector_type(4)));
typedef int          v8i __attribute__((ext_vector_type(8)));
typedef int          v4i __attribute__((ext_vector_type(4)));

#if defined(__has_builtin)
#if __has_builtin(__builtin_amdgcn_tensor_load_to_lds) && \
    __has_builtin(__builtin_amdgcn_s_wait_tensorcnt)
#define HAVE_TDM 1
#endif
#endif
#ifndef HAVE_TDM
#define HAVE_TDM 0
#endif

union Frag { v16h h; unsigned int u[8]; };

__device__ __forceinline__ v8f wmma16(const Frag &a, const Frag &b, v8f c) {
  // 8 args: (neg_a, A, neg_b, B, c_mod, C, reuse_a, reuse_b)
  return __builtin_amdgcn_wmma_f32_16x16x32_f16(false, a.h, false, b.h,
                                                (short)0, c, false, false);
}

#if HAVE_TDM
// ---------------------------------------------------------------------------
// TDM: DMA a (rows x 32) f16 tile (row stride = strideElems halfs) from
// global into LDS at byte offset lds_off, inserting 1 DWORD (4 B) of padding
// after every 16 DWORDs (64 B) -> matches LDS row stride of 34 halfwords.
// D# bit layout per CDNA5 ISA 8.3/8.4 (group0 128b, group1 256b; 2D tensor,
// groups 2/3 unused -> zero). This toolchain exposes the 6-arg builtin:
// (uint32x4 g0, int32x8 g1, int32x4 g2, int32x4 g3, int32x8, i32 cpol).
// ---------------------------------------------------------------------------
__device__ __forceinline__ void tdm_load_tile_f16(
    const half_t *gsrc, unsigned lds_off, unsigned rows, unsigned kElems,
    unsigned strideElems, unsigned tensorRows)
{
  unsigned long long ga = (unsigned long long)(uintptr_t)gsrc;
  v4u g0;
  g0[0] = 1u;                                   // count=1 (valid), user desc
  g0[1] = lds_off;                              // lds_addr (bytes)
  g0[2] = (unsigned)(ga & 0xFFFFFFFFu);         // global_addr[31:0]
  g0[3] = (unsigned)((ga >> 32) & 0x01FFFFFFu)  // global_addr[56:32]
          | (2u << 30);                         // type=2 ("image")
  v8i g1;
  g1[0] = (int)((1u << 16)     // data_size = 1 -> 2 bytes
              | (1u << 20)     // pad_enable
              | (3u << 22));   // pad_interval = 3 -> 16 DWORDs (pad_amount=0 -> 1 DWORD)
  g1[1] = (int)((kElems & 0xFFFFu) << 16);                    // tensor_dim0[15:0]
  g1[2] = (int)(((kElems >> 16) & 0xFFFFu)                    // tensor_dim0[31:16]
              | ((tensorRows & 0xFFFFu) << 16));              // tensor_dim1[15:0]
  g1[3] = (int)(((tensorRows >> 16) & 0xFFFFu)                // tensor_dim1[31:16]
              | (32u << 16));                                 // tile_dim0 = 32 elems
  g1[4] = (int)(rows & 0xFFFFu);                              // tile_dim1; tile_dim2=0
  g1[5] = (int)strideElems;                                   // tensor_dim0_stride[31:0]
  g1[6] = 0;                                                  // stride hi / dim1_stride lo
  g1[7] = 0;
  v4i z4 = {0, 0, 0, 0};
  v8i z8 = {0, 0, 0, 0, 0, 0, 0, 0};
  __builtin_amdgcn_tensor_load_to_lds(g0, g1, z4, z4, z8, 0);
}
#endif

// ---------------------------------------------------------------------------
// Generic WMMA GEMM:  Out = act(A @ W + bias) [+ Res]
// A: MxK row-major (AT = float | _Float16), W: KxN row-major fp32,
// Out: MxN (OT), accumulation fp32. gelu!=0 -> exact GELU before residual.
// Tile: 64x128 per 256-thread block; 8 waves, each wave a 16x64 strip.
// ---------------------------------------------------------------------------
template <typename AT, typename OT>
__global__ __launch_bounds__(256) void gemm_wmma(
    const AT *__restrict__ A, const float *__restrict__ W,
    const float *__restrict__ bias, OT *__restrict__ Out,
    const float *__restrict__ Res, int M, int K, int N, int gelu)
{
  constexpr int BM = 64, BN = 128, BK = 32, SK = BK + 2;  // pad: 34 halves
  __shared__ half_t lA[BM * SK];   // [m][k]
  __shared__ half_t lB[BN * SK];   // [n][k]  (W transposed into LDS)
  const int tid = threadIdx.x, lane = tid & 31, wid = tid >> 5;
  const int mBase = blockIdx.y * BM, nBase = blockIdx.x * BN;
  const int wr = (wid & 3) * 16;   // wave row offset in tile
  const int wc = (wid >> 2) * 64;  // wave col offset in tile
  const int hw = lane >> 4, lr = lane & 15;
  v8f acc[4] = {};

  for (int k0 = 0; k0 < K; k0 += BK) {
    // stage A slab 64x32 (fp -> f16)
    for (int i = tid; i < BM * BK; i += 256) {
      int m = i >> 5, kk = i & 31;
      int gm = mBase + m, gk = k0 + kk;
      float v = (gm < M && gk < K) ? (float)A[(size_t)gm * K + gk] : 0.0f;
      lA[m * SK + kk] = (half_t)v;
    }
    // stage W slab 32x128, transposed to [n][k]
    for (int i = tid; i < BK * BN; i += 256) {
      int kk = i >> 7, n = i & 127;
      int gk = k0 + kk, gn = nBase + n;
      float v = (gk < K && gn < N) ? W[(size_t)gk * N + gn] : 0.0f;
      lB[n * SK + kk] = (half_t)v;
    }
    if (tid == 0 && k0 + BK < K)   // emits global_prefetch_b8
      __builtin_prefetch(&W[(size_t)(k0 + BK) * N + nBase], 0, 3);
    __syncthreads();

    // A fragment (16x32, contiguous k = 16*hw + s: packed K-pairs)
    Frag af;
#pragma unroll
    for (int v = 0; v < 8; ++v) {
      int kk = hw * 16 + v * 2;
      af.u[v] = *(const unsigned int *)&lA[(wr + lr) * SK + kk];
    }
#pragma unroll
    for (int t = 0; t < 4; ++t) {
      // B fragment (32x16, contiguous k = 16*hw + s, same as A)
      Frag bf;
#pragma unroll
      for (int v = 0; v < 8; ++v) {
        int kk = hw * 16 + v * 2;
        bf.u[v] = *(const unsigned int *)&lB[(wc + t * 16 + lr) * SK + kk];
      }
      acc[t] = wmma16(af, bf, acc[t]);
    }
    __syncthreads();
  }

  // epilogue: C/D layout -> element (M = i + 8*(lane/16), N = lane%16)
#pragma unroll
  for (int t = 0; t < 4; ++t) {
#pragma unroll
    for (int i = 0; i < 8; ++i) {
      int m = mBase + wr + i + hw * 8;
      int n = nBase + wc + t * 16 + lr;
      if (m < M && n < N) {
        float v = acc[t][i] + bias[n];
        if (gelu) v = 0.5f * v * (1.0f + erff(v * 0.70710678118654752f));
        if (Res) v += Res[(size_t)m * N + n];
        Out[(size_t)m * N + n] = (OT)v;
      }
    }
  }
}

// ---------------------------------------------------------------------------
// WMMA GEMM with TDM A-staging:  Out(f16) = A(f16) @ W(f32) + bias
// Used for the relation-encoder layers (M=524288, K=256): A is already f16
// in global, so the Tensor Data Mover DMAs each 64x32 tile directly into
// padded LDS (no VGPR round-trip). Requires M%64==0 and K%32==0.
// ---------------------------------------------------------------------------
__global__ __launch_bounds__(256) void gemm_wmma_tdm(
    const half_t *__restrict__ A, const float *__restrict__ W,
    const float *__restrict__ bias, half_t *__restrict__ Out,
    int M, int K, int N)
{
  constexpr int BM = 64, BN = 128, BK = 32, SK = BK + 2;
  __shared__ half_t lA[BM * SK];   // [m][k], row stride 68 B (64 B + 4 B pad)
  __shared__ half_t lB[BN * SK];   // [n][k]
  const int tid = threadIdx.x, lane = tid & 31, wid = tid >> 5;
  const int mBase = blockIdx.y * BM, nBase = blockIdx.x * BN;
  const int wr = (wid & 3) * 16;
  const int wc = (wid >> 2) * 64;
  const int hw = lane >> 4, lr = lane & 15;
  v8f acc[4] = {};
#if HAVE_TDM
  const unsigned ldsA_off = (unsigned)(uintptr_t)&lA[0];
#endif

  for (int k0 = 0; k0 < K; k0 += BK) {
#if HAVE_TDM
    if (wid == 0) {
      tdm_load_tile_f16(A + (size_t)mBase * K + k0, ldsA_off,
                        (unsigned)BM, (unsigned)K, (unsigned)K, (unsigned)M);
      __builtin_amdgcn_s_wait_tensorcnt(0);   // s_wait_tensorcnt 0
    }
#else
    for (int i = tid; i < BM * BK; i += 256) {
      int m = i >> 5, kk = i & 31;
      lA[m * SK + kk] = A[(size_t)(mBase + m) * K + k0 + kk];
    }
#endif
    for (int i = tid; i < BK * BN; i += 256) {
      int kk = i >> 7, n = i & 127;
      int gk = k0 + kk, gn = nBase + n;
      float v = (gn < N) ? W[(size_t)gk * N + gn] : 0.0f;
      lB[n * SK + kk] = (half_t)v;
    }
    __syncthreads();   // publishes TDM-written lA + staged lB to all waves

    Frag af;
#pragma unroll
    for (int v = 0; v < 8; ++v) {
      int kk = hw * 16 + v * 2;
      af.u[v] = *(const unsigned int *)&lA[(wr + lr) * SK + kk];
    }
#pragma unroll
    for (int t = 0; t < 4; ++t) {
      Frag bf;
#pragma unroll
      for (int v = 0; v < 8; ++v) {
        int kk = hw * 16 + v * 2;
        bf.u[v] = *(const unsigned int *)&lB[(wc + t * 16 + lr) * SK + kk];
      }
      acc[t] = wmma16(af, bf, acc[t]);
    }
    __syncthreads();
  }

#pragma unroll
  for (int t = 0; t < 4; ++t) {
#pragma unroll
    for (int i = 0; i < 8; ++i) {
      int m = mBase + wr + i + hw * 8;
      int n = nBase + wc + t * 16 + lr;
      if (n < N)
        Out[(size_t)m * N + n] = (half_t)(acc[t][i] + bias[n]);
    }
  }
}

// ---------------------------------------------------------------------------
// LayerNorm over C=128, one wave32 per row (4 elems/lane), fp32 in/out.
// ---------------------------------------------------------------------------
__global__ __launch_bounds__(256) void layernorm128(
    const float *__restrict__ x, const float *__restrict__ g,
    const float *__restrict__ b, float *__restrict__ y, int rows)
{
  int row = blockIdx.x * 8 + (threadIdx.x >> 5);
  int lane = threadIdx.x & 31;
  if (row >= rows) return;
  const float *px = x + (size_t)row * 128;
  float v[4], s = 0.f;
#pragma unroll
  for (int i = 0; i < 4; ++i) { v[i] = px[lane + i * 32]; s += v[i]; }
#pragma unroll
  for (int o = 16; o >= 1; o >>= 1) s += __shfl_xor(s, o, 32);
  float mu = s * (1.0f / 128.0f), q = 0.f;
#pragma unroll
  for (int i = 0; i < 4; ++i) { float d = v[i] - mu; q += d * d; }
#pragma unroll
  for (int o = 16; o >= 1; o >>= 1) q += __shfl_xor(q, o, 32);
  float rstd = rsqrtf(q * (1.0f / 128.0f) + 1e-5f);
  float *py = y + (size_t)row * 128;
#pragma unroll
  for (int i = 0; i < 4; ++i) {
    int c = lane + i * 32;
    py[c] = (v[i] - mu) * rstd * g[c] + b[c];
  }
}

// ---------------------------------------------------------------------------
// Fused LN2 + rconv (128 -> 8):  attn[b,h,n,m] = LN(rf[b,n,m,:]) @ Wc + bc
// One wave32 per (b,n,m) row. rf stored f16, math fp32.
// ---------------------------------------------------------------------------
__global__ __launch_bounds__(256) void attn_r_ln(
    const half_t *__restrict__ rf, const float *__restrict__ g,
    const float *__restrict__ bb, const float *__restrict__ Wc,
    const float *__restrict__ bc, float *__restrict__ attn,
    int N, int H)
{
  size_t row = (size_t)blockIdx.x * 8 + (threadIdx.x >> 5);
  int lane = threadIdx.x & 31;
  const half_t *x = rf + row * 128;
  float v[4], s = 0.f;
#pragma unroll
  for (int i = 0; i < 4; ++i) { v[i] = (float)x[lane + i * 32]; s += v[i]; }
#pragma unroll
  for (int o = 16; o >= 1; o >>= 1) s += __shfl_xor(s, o, 32);
  float mu = s * (1.0f / 128.0f), q = 0.f;
#pragma unroll
  for (int i = 0; i < 4; ++i) { float d = v[i] - mu; q += d * d; }
#pragma unroll
  for (int o = 16; o >= 1; o >>= 1) q += __shfl_xor(q, o, 32);
  float rstd = rsqrtf(q * (1.0f / 128.0f) + 1e-5f);
  float acc[8] = {0, 0, 0, 0, 0, 0, 0, 0};
#pragma unroll
  for (int i = 0; i < 4; ++i) {
    int c = lane + i * 32;
    float y = (v[i] - mu) * rstd * g[c] + bb[c];
#pragma unroll
    for (int hh = 0; hh < 8; ++hh) acc[hh] += y * Wc[c * 8 + hh];
  }
#pragma unroll
  for (int o = 16; o >= 1; o >>= 1)
#pragma unroll
    for (int hh = 0; hh < 8; ++hh) acc[hh] += __shfl_xor(acc[hh], o, 32);
  int m = (int)(row % N);
  int n = (int)((row / N) % N);
  int b = (int)(row / ((size_t)N * N));
  float outv = 0.f;
#pragma unroll
  for (int hh = 0; hh < 8; ++hh) if (lane == hh) outv = acc[hh] + bc[hh];
  if (lane < 8)
    attn[(((size_t)(b * H + lane)) * N + n) * N + m] = outv;
}

// ---------------------------------------------------------------------------
// attn = (q @ k^T + R) * conn * SCALE   per (b,h) workgroup.
// q,k staged to LDS as f16, d=16 zero-padded to K=32 for one WMMA step.
// attn buffer already holds R on entry; overwritten with pre-softmax scores.
// ---------------------------------------------------------------------------
__global__ __launch_bounds__(256) void attn_qk(
    const float *__restrict__ qkv, const float *__restrict__ conn,
    float *__restrict__ attn)
{
  constexpr int SK = 34;
  __shared__ half_t lQ[256 * SK];  // [n][d]
  __shared__ half_t lK[256 * SK];  // [m][d]  (B-operand [n][k] layout)
  const int bh = blockIdx.x, b = bh >> 3, h = bh & 7;
  const int tid = threadIdx.x, lane = tid & 31, wid = tid >> 5;
  for (int i = tid; i < 256 * 32; i += 256) {
    int n = i >> 5, d = i & 31;
    size_t base = (size_t)(b * 256 + n) * 384 + h * 16;
    float qv = (d < 16) ? qkv[base + d] : 0.f;         // j=0 (q)
    float kv = (d < 16) ? qkv[base + 128 + d] : 0.f;   // j=1 (k)
    lQ[n * SK + d] = (half_t)qv;
    lK[n * SK + d] = (half_t)kv;
  }
  __syncthreads();
  const int hw = lane >> 4, lr = lane & 15;
#pragma unroll
  for (int trr = 0; trr < 2; ++trr) {
    int tr = wid * 2 + trr;
    Frag af;
#pragma unroll
    for (int v = 0; v < 8; ++v) {
      int kk = hw * 16 + v * 2;
      af.u[v] = *(const unsigned int *)&lQ[(tr * 16 + lr) * SK + kk];
    }
    for (int tc = 0; tc < 16; ++tc) {
      Frag bf;
#pragma unroll
      for (int v = 0; v < 8; ++v) {
        int kk = hw * 16 + v * 2;
        bf.u[v] = *(const unsigned int *)&lK[(tc * 16 + lr) * SK + kk];
      }
      v8f c = {};
      c = wmma16(af, bf, c);
#pragma unroll
      for (int i = 0; i < 8; ++i) {
        int n = tr * 16 + i + hw * 8;
        int m = tc * 16 + lr;
        size_t idx = ((size_t)bh * 256 + n) * 256 + m;
        float R = attn[idx];
        float cn = conn[((size_t)b * 256 + n) * 256 + m];
        attn[idx] = (c[i] + R) * cn * 0.25f;
      }
    }
  }
}

// ---------------------------------------------------------------------------
// Row softmax over L=256, one wave32 per row.
// ---------------------------------------------------------------------------
__global__ __launch_bounds__(256) void softmax_rows(float *__restrict__ attn,
                                                    int rows)
{
  int row = blockIdx.x * 8 + (threadIdx.x >> 5);
  int lane = threadIdx.x & 31;
  if (row >= rows) return;
  float *p = attn + (size_t)row * 256;
  float x[8], mx = -3.4e38f;
#pragma unroll
  for (int i = 0; i < 8; ++i) { x[i] = p[lane + i * 32]; mx = fmaxf(mx, x[i]); }
#pragma unroll
  for (int o = 16; o >= 1; o >>= 1) mx = fmaxf(mx, __shfl_xor(mx, o, 32));
  float s = 0.f;
#pragma unroll
  for (int i = 0; i < 8; ++i) { x[i] = expf(x[i] - mx); s += x[i]; }
#pragma unroll
  for (int o = 16; o >= 1; o >>= 1) s += __shfl_xor(s, o, 32);
  float inv = 1.0f / s;
#pragma unroll
  for (int i = 0; i < 8; ++i) p[lane + i * 32] = x[i] * inv;
}

// ---------------------------------------------------------------------------
// o[b,n,h*16+d] = attn @ v   per (b,h) workgroup, WMMA with K-loop over 256.
// ---------------------------------------------------------------------------
__global__ __launch_bounds__(256) void attn_pv(
    const float *__restrict__ attn, const float *__restrict__ qkv,
    float *__restrict__ o)
{
  constexpr int SK = 34, SV = 272;
  __shared__ half_t lP[256 * SK];  // [n][k-slab]
  __shared__ half_t lV[16 * SV];   // [d][m]  (B-operand [n][k] layout)
  const int bh = blockIdx.x, b = bh >> 3, h = bh & 7;
  const int tid = threadIdx.x, lane = tid & 31, wid = tid >> 5;
  for (int i = tid; i < 256 * 16; i += 256) {
    int m = i >> 4, d = i & 15;
    lV[d * SV + m] =
        (half_t)qkv[(size_t)(b * 256 + m) * 384 + 256 + h * 16 + d];  // j=2 (v)
  }
  const int hw = lane >> 4, lr = lane & 15;
  v8f acc[2] = {};
  for (int k0 = 0; k0 < 256; k0 += 32) {
    __syncthreads();
    for (int i = tid; i < 256 * 32; i += 256) {
      int n = i >> 5, kk = i & 31;
      lP[n * SK + kk] = (half_t)attn[((size_t)bh * 256 + n) * 256 + k0 + kk];
    }
    __syncthreads();
#pragma unroll
    for (int t = 0; t < 2; ++t) {
      int tr = wid * 2 + t;
      Frag af;
#pragma unroll
      for (int v = 0; v < 8; ++v) {
        int kk = hw * 16 + v * 2;
        af.u[v] = *(const unsigned int *)&lP[(tr * 16 + lr) * SK + kk];
      }
      Frag bf;
#pragma unroll
      for (int v = 0; v < 8; ++v) {
        int kk = hw * 16 + v * 2;
        bf.u[v] = *(const unsigned int *)&lV[lr * SV + k0 + kk];
      }
      acc[t] = wmma16(af, bf, acc[t]);
    }
  }
#pragma unroll
  for (int t = 0; t < 2; ++t) {
    int tr = wid * 2 + t;
#pragma unroll
    for (int i = 0; i < 8; ++i) {
      int n = tr * 16 + i + hw * 8;
      o[(size_t)(b * 256 + n) * 128 + h * 16 + lr] = acc[t][i];
    }
  }
}

// ---------------------------------------------------------------------------
// Host orchestration
// ---------------------------------------------------------------------------
extern "C" void kernel_launch(void *const *d_in, const int *in_sizes, int n_in,
                              void *d_out, int out_size, void *d_ws,
                              size_t ws_size, hipStream_t stream)
{
  (void)in_sizes; (void)n_in; (void)out_size; (void)ws_size;
  const int B = 8, N = 256, C = 128, H = 8, DEPTH = 4;
  const int ROWS = B * N;                       // 2048
  const size_t RREL = (size_t)B * N * N;        // 524288

  const float *joint_in    = (const float *)d_in[0];
  const float *relation_in = (const float *)d_in[1];
  const float *conn        = (const float *)d_in[2];
  const float *je_w1 = (const float *)d_in[3],  *je_b1 = (const float *)d_in[4];
  const float *je_w2 = (const float *)d_in[5],  *je_b2 = (const float *)d_in[6];
  const float *je_w3 = (const float *)d_in[7],  *je_b3 = (const float *)d_in[8];
  const float *re_w1 = (const float *)d_in[9],  *re_b1 = (const float *)d_in[10];
  const float *re_w2 = (const float *)d_in[11], *re_b2 = (const float *)d_in[12];
  const float *re_w3 = (const float *)d_in[13], *re_b3 = (const float *)d_in[14];
  const float *qkv_w = (const float *)d_in[15], *qkv_b = (const float *)d_in[16];
  const float *rconv_w = (const float *)d_in[17], *rconv_b = (const float *)d_in[18];
  const float *proj_w = (const float *)d_in[19], *proj_b = (const float *)d_in[20];
  const float *ln1_g = (const float *)d_in[21], *ln1_b = (const float *)d_in[22];
  const float *ln2_g = (const float *)d_in[23], *ln2_b = (const float *)d_in[24];
  const float *ln3_g = (const float *)d_in[25], *ln3_b = (const float *)d_in[26];
  const float *mw1 = (const float *)d_in[27], *mb1 = (const float *)d_in[28];
  const float *mw2 = (const float *)d_in[29], *mb2 = (const float *)d_in[30];
  const float *ng  = (const float *)d_in[31], *nb  = (const float *)d_in[32];
  const float *dw1 = (const float *)d_in[33], *db1 = (const float *)d_in[34];
  const float *dw2 = (const float *)d_in[35], *db2 = (const float *)d_in[36];
  const float *dw3 = (const float *)d_in[37], *db3 = (const float *)d_in[38];

  char *ws = (char *)d_ws;
  const size_t MB = 1ull << 20;
  float  *jf   = (float *)(ws + 0 * MB);     // (B*N,128)  1 MB
  float  *xln  = (float *)(ws + 1 * MB);     // 1 MB
  float  *hln  = (float *)(ws + 2 * MB);     // 1 MB
  float  *obuf = (float *)(ws + 3 * MB);     // 1 MB
  float  *mlp1 = (float *)(ws + 4 * MB);     // 1 MB
  float  *qkvb = (float *)(ws + 5 * MB);     // (B*N,384)  3 MB
  float  *t1   = (float *)(ws + 8 * MB);     // (B*N,256)  2 MB
  float  *t2   = (float *)(ws + 10 * MB);    // (B*N,512)  4 MB
  float  *attn = (float *)(ws + 16 * MB);    // (B,H,N,N)  16 MB
  half_t *rfh  = (half_t *)(ws + 32 * MB);   // (B,N,N,128) f16  128 MB
  half_t *h1   = (half_t *)(ws + 160 * MB);  // (B,N,N,256) f16  256 MB
  half_t *h2   = (half_t *)(ws + 416 * MB);  // (B,N,N,256) f16  256 MB

  auto ggrid = [](int Mm, int Nn) {
    return dim3((unsigned)((Nn + 127) / 128), (unsigned)((Mm + 63) / 64));
  };

  // ---- joint encoder: 96 -> 256 -> 256 -> 128 ----
  gemm_wmma<float, float><<<ggrid(ROWS, 256), 256, 0, stream>>>(
      joint_in, je_w1, je_b1, t1, nullptr, ROWS, 96, 256, 0);
  gemm_wmma<float, float><<<ggrid(ROWS, 256), 256, 0, stream>>>(
      t1, je_w2, je_b2, t2, nullptr, ROWS, 256, 256, 0);
  gemm_wmma<float, float><<<ggrid(ROWS, 128), 256, 0, stream>>>(
      t2, je_w3, je_b3, jf, nullptr, ROWS, 256, 128, 0);

  // ---- relation encoder: 26 -> 256 -> 256 -> 128, f16 intermediates ----
  // layer 1: fp32 A (K=26) -> classic staging path
  gemm_wmma<float, half_t><<<ggrid((int)RREL, 256), 256, 0, stream>>>(
      relation_in, re_w1, re_b1, h1, nullptr, (int)RREL, 26, 256, 0);
  // layers 2/3: f16 A, M%64==0, K%32==0 -> TDM direct-to-LDS staging
  gemm_wmma_tdm<<<ggrid((int)RREL, 256), 256, 0, stream>>>(
      h1, re_w2, re_b2, h2, (int)RREL, 256, 256);
  gemm_wmma_tdm<<<ggrid((int)RREL, 128), 256, 0, stream>>>(
      h2, re_w3, re_b3, rfh, (int)RREL, 256, 128);

  // ---- transformer blocks ----
  for (int i = 0; i < DEPTH; ++i) {
    layernorm128<<<ROWS / 8, 256, 0, stream>>>(jf, ln1_g + i * C, ln1_b + i * C,
                                               xln, ROWS);
    gemm_wmma<float, float><<<ggrid(ROWS, 3 * C), 256, 0, stream>>>(
        xln, qkv_w + (size_t)i * C * 3 * C, qkv_b + i * 3 * C, qkvb, nullptr,
        ROWS, C, 3 * C, 0);
    attn_r_ln<<<(unsigned)(RREL / 8), 256, 0, stream>>>(
        rfh, ln2_g + i * C, ln2_b + i * C, rconv_w + (size_t)i * C * H,
        rconv_b + i * H, attn, N, H);
    attn_qk<<<B * H, 256, 0, stream>>>(qkvb, conn, attn);
    softmax_rows<<<(B * H * N) / 8, 256, 0, stream>>>(attn, B * H * N);
    attn_pv<<<B * H, 256, 0, stream>>>(attn, qkvb, obuf);
    gemm_wmma<float, float><<<ggrid(ROWS, C), 256, 0, stream>>>(
        obuf, proj_w + (size_t)i * C * C, proj_b + i * C, jf, jf, ROWS, C, C, 0);
    layernorm128<<<ROWS / 8, 256, 0, stream>>>(jf, ln3_g + i * C, ln3_b + i * C,
                                               hln, ROWS);
    gemm_wmma<float, float><<<ggrid(ROWS, C), 256, 0, stream>>>(
        hln, mw1 + (size_t)i * C * C, mb1 + i * C, mlp1, nullptr, ROWS, C, C, 1);
    gemm_wmma<float, float><<<ggrid(ROWS, C), 256, 0, stream>>>(
        mlp1, mw2 + (size_t)i * C * C, mb2 + i * C, jf, jf, ROWS, C, C, 0);
  }

  // ---- final LN + decoder 128 -> 256 -> 512 -> 90 ----
  layernorm128<<<ROWS / 8, 256, 0, stream>>>(jf, ng, nb, xln, ROWS);
  gemm_wmma<float, float><<<ggrid(ROWS, 256), 256, 0, stream>>>(
      xln, dw1, db1, t1, nullptr, ROWS, 128, 256, 0);
  gemm_wmma<float, float><<<ggrid(ROWS, 512), 256, 0, stream>>>(
      t1, dw2, db2, t2, nullptr, ROWS, 256, 512, 0);
  gemm_wmma<float, float><<<ggrid(ROWS, 90), 256, 0, stream>>>(
      t2, dw3, db3, (float *)d_out, nullptr, ROWS, 512, 90, 0);
}
